// ViTIntermediate_8117488190192
// MI455X (gfx1250) — 8-bit matrix operand order repaired and checked by reading, compiled, not run on hardware
//
#include <hip/hip_runtime.h>
#include <hip/hip_bf16.h>

typedef __attribute__((ext_vector_type(8))) int v8i;

#define HDIM 768
#define IDIM 3072
#define KSTEPS 12
#define TILES_PER_WAVE 12
#define WAVES 16
#define BLOCK 512

__device__ __forceinline__ signed char sgn8(float v) {
    return v > 0.f ? (signed char)1 : (v < 0.f ? (signed char)-1 : (signed char)0);
}

__global__ void pack_w_kernel(const float* __restrict__ W,
                              signed char* __restrict__ wq,
                              float* __restrict__ alpha) {
    __shared__ float red[256];
    const int i = blockIdx.x;
    const int t = threadIdx.x;
    float s = 0.f;
    for (int h = t; h < HDIM; h += 256) {
        float w = W[(size_t)i * HDIM + h];
        s += fabsf(w);
        wq[(size_t)i * HDIM + h] = sgn8(w);
    }
    red[t] = s;
    __syncthreads();
    for (int off = 128; off > 0; off >>= 1) {
        if (t < off) red[t] += red[t + off];
        __syncthreads();
    }
    if (t == 0) alpha[i] = red[0] * (1.0f / (float)HDIM);
}

__global__ void pack_x_kernel(const float* __restrict__ x,
                              const float* __restrict__ move,
                              signed char* __restrict__ xq,
                              int nquads) {
    int f = blockIdx.x * blockDim.x + threadIdx.x;
    if (f >= nquads) return;
    int base = f * 4;
    int h = base % HDIM;
    float4 xv = *(const float4*)(x + base);
    char4 r;
    r.x = sgn8(xv.x + move[h + 0]);
    r.y = sgn8(xv.y + move[h + 1]);
    r.z = sgn8(xv.z + move[h + 2]);
    r.w = sgn8(xv.w + move[h + 3]);
    *(char4*)(xq + base) = r;
}

__global__ __launch_bounds__(BLOCK)
void bin_gemm_ln_kernel(const signed char* __restrict__ xq,
                        const signed char* __restrict__ wq,
                        const float* __restrict__ alpha,
                        const float* __restrict__ bias,
                        const float* __restrict__ gamma,
                        const float* __restrict__ beta,
                        const float* __restrict__ move1,
                        const float* __restrict__ pal,
                        const float* __restrict__ move2,
                        const float* __restrict__ hidden,
                        float* __restrict__ out) {
    __shared__ __align__(16) signed char smemA[16 * HDIM];
    __shared__ float redS[16 * WAVES];
    __shared__ float redQ[16 * WAVES];
    __shared__ float stats[16 * 2];

    const int t    = threadIdx.x;
    const int lane = t & 31;
    const int wave = t >> 5;
    const int m0   = blockIdx.x * 16;
    const int l15  = lane & 15;
    const int half = lane >> 4;

    {
        const int4* src = (const int4*)(xq + (size_t)m0 * HDIM);
        int4* dst = (int4*)smemA;
        for (int c = t; c < (16 * HDIM) / 16; c += BLOCK) dst[c] = src[c];
    }
    __syncthreads();

    v8i acc[TILES_PER_WAVE] = {};
    const int colBase = wave * (TILES_PER_WAVE * 16);

    for (int ks = 0; ks < KSTEPS; ++ks) {
        union { v8i v; unsigned long long qw[4]; } ua;
        const signed char* abase = smemA + l15 * HDIM + ks * 64 + half * 16;
        #pragma unroll
        for (int c = 0; c < 4; ++c)
            ua.qw[c] = *(const unsigned long long*)(abase + (c & 1) * 8 + (c >> 1) * 32);

        #pragma unroll
        for (int tt = 0; tt < TILES_PER_WAVE; ++tt) {
            const signed char* bbase =
                wq + (size_t)(colBase + tt * 16 + l15) * HDIM + ks * 64 + half * 16;
            union { v8i v; int4 x2[2]; } ub;
            ub.x2[0] = *(const int4*)(bbase);
            ub.x2[1] = *(const int4*)(bbase + 32);
            acc[tt] = __builtin_amdgcn_wmma_i32_16x16x64_iu8(
                true, ua.v, true, ub.v, acc[tt], false, false);
        }
    }

    float s[8] = {0, 0, 0, 0, 0, 0, 0, 0};
    float q[8] = {0, 0, 0, 0, 0, 0, 0, 0};
    #pragma unroll
    for (int tt = 0; tt < TILES_PER_WAVE; ++tt) {
        const int i = colBase + tt * 16 + l15;
        const float al = alpha[i];
        const float bi = bias[i];
        #pragma unroll
        for (int r = 0; r < 8; ++r) {
            float yf = al * (float)acc[tt][r] + bi;
            s[r] += yf;
            q[r] += yf * yf;
        }
    }
    #pragma unroll
    for (int r = 0; r < 8; ++r) {
        float ss = s[r], qq = q[r];
        #pragma unroll
        for (int m = 1; m <= 8; m <<= 1) {
            ss += __shfl_xor(ss, m, 32);
            qq += __shfl_xor(qq, m, 32);
        }
        if (l15 == 0) {
            int row = half * 8 + r;
            redS[row * WAVES + wave] = ss;
            redQ[row * WAVES + wave] = qq;
        }
    }
    __syncthreads();

    if (lane < WAVES) {
        float ss = redS[wave * WAVES + lane];
        float qq = redQ[wave * WAVES + lane];
        #pragma unroll
        for (int m = 1; m <= 8; m <<= 1) {
            ss += __shfl_xor(ss, m, 32);
            qq += __shfl_xor(qq, m, 32);
        }
        if (lane == 0) {
            float mean = ss * (1.0f / (float)IDIM);
            float var  = qq * (1.0f / (float)IDIM) - mean * mean;
            stats[wave * 2 + 0] = mean;
            stats[wave * 2 + 1] = rsqrtf(var + 1e-12f);
        }
    }
    __syncthreads();

    #pragma unroll
    for (int tt = 0; tt < TILES_PER_WAVE; ++tt) {
        const int i = colBase + tt * 16 + l15;
        const float al = alpha[i];
        const float bi = bias[i];
        const float g  = gamma[i];
        const float be = beta[i];
        const float m1 = move1[i];
        const float pa = pal[i];
        const float m2 = move2[i];
        const int hcol = i % HDIM;
        #pragma unroll
        for (int r = 0; r < 8; ++r) {
            const int row = half * 8 + r;
            const int gm  = m0 + row;
            float yf = al * (float)acc[tt][r] + bi;
            float yn = (yf - stats[row * 2]) * stats[row * 2 + 1] * g + be;
            float o  = yn + hidden[(size_t)gm * HDIM + hcol];
            float z  = o - m1;
            o = (z >= 0.f ? z : pa * z) + m2;
            out[(size_t)gm * IDIM + i] = o;
        }
    }
}

extern "C" void kernel_launch(void* const* d_in, const int* in_sizes, int n_in,
                              void* d_out, int out_size, void* d_ws, size_t ws_size,
                              hipStream_t stream) {
    const float* hs    = (const float*)d_in[0];
    const float* W     = (const float*)d_in[1];
    const float* bias  = (const float*)d_in[2];
    const float* move  = (const float*)d_in[3];
    const float* gamma = (const float*)d_in[4];
    const float* beta  = (const float*)d_in[5];
    const float* move1 = (const float*)d_in[6];
    const float* pal   = (const float*)d_in[7];
    const float* move2 = (const float*)d_in[8];
    float* out = (float*)d_out;

    const int M = in_sizes[0] / HDIM;

    signed char* wq = (signed char*)d_ws;
    signed char* xq = wq + (size_t)IDIM * HDIM;
    float* alpha = (float*)(xq + (size_t)M * HDIM);

    pack_w_kernel<<<IDIM, 256, 0, stream>>>(W, wq, alpha);

    const int nquads = (M * HDIM) / 4;
    pack_x_kernel<<<(nquads + 255) / 256, 256, 0, stream>>>(hs, move, xq, nquads);

    bin_gemm_ln_kernel<<<M / 16, BLOCK, 0, stream>>>(
        xq, wq, alpha, bias, gamma, beta, move1, pal, move2, hs, out);
}
